// LSTM_17600775979701
// MI455X (gfx1250) — matrix operand order repaired and checked by reading, not compiled, not run on hardware
//
#include <hip/hip_runtime.h>
#include <cmath>

typedef _Float16 f16;
typedef __attribute__((ext_vector_type(16))) _Float16 v16h;
typedef __attribute__((ext_vector_type(8)))  _Float16 v8h;
typedef __attribute__((ext_vector_type(8)))  float    v8f;

#define B_   1024
#define S_   512
#define H_   128
#define G4H_ 512
#define C_   10

// workspace layout (bytes)
#define WS_WH_OFF 0                   // f16[128*512] = 131072 B, WMMA-B swizzled
#define WS_WX_OFF (131072)            // float[512]
#define WS_B_OFF  (131072 + 2048)     // float[512]

// LDS layout (bytes): x tile (transposed) + ping-pong h tiles + final f32 h
#define L_X    0                       // f32 [512][16]  (32768)
#define L_H0   (32768)                 // f16 [16][128]  ( 4096)
#define L_H1   (32768 + 4096)          // f16 [16][128]  ( 4096)
#define L_HF   (32768 + 8192)          // f32 [16][128]  ( 8192)
#define SMEM_BYTES (32768 + 8192 + 8192)   // 49152

// fast activations (saturate correctly at +/-inf)
__device__ __forceinline__ float fast_sigmoid(float v) {
  return __builtin_amdgcn_rcpf(1.f + __expf(-v));
}
__device__ __forceinline__ float fast_tanh(float v) {
  return 1.f - 2.f * __builtin_amdgcn_rcpf(__expf(2.f * v) + 1.f);
}

// ---------------------------------------------------------------------------
// Prep: build f16 Wh in WMMA-B fragment order + concatenated Wx / bias.
// Swizzle: flat idx = ((ntile*4 + kt)*32 + lane)*16 + e
//   holds Wh_cat[ kt*32 + (lane>>4)*16 + e ][ ntile*16 + (lane&15) ]
// so a B fragment for (ntile,kt) is 16 contiguous halves per lane, and
// consecutive lanes read consecutive 32B chunks (coalesced).
// ---------------------------------------------------------------------------
__global__ void lstm_prep(const float* __restrict__ Wgh, const float* __restrict__ Wih,
                          const float* __restrict__ Wfh, const float* __restrict__ Woh,
                          const float* __restrict__ Wgx, const float* __restrict__ Wix,
                          const float* __restrict__ Wfx, const float* __restrict__ Wox,
                          const float* __restrict__ bg,  const float* __restrict__ bi,
                          const float* __restrict__ bf,  const float* __restrict__ bo,
                          f16* __restrict__ wh_swz, float* __restrict__ wx_cat,
                          float* __restrict__ b_cat) {
  int idx = blockIdx.x * blockDim.x + threadIdx.x;
  if (idx < H_ * G4H_) {
    int e     = idx & 15;
    int lane  = (idx >> 4) & 31;
    int kt    = (idx >> 9) & 3;
    int ntile = idx >> 11;                  // 0..31
    int ncol  = ntile * 16 + (lane & 15);   // 0..511 (gate-concatenated col)
    int k     = kt * 32 + (lane >> 4) * 16 + e;
    int gate  = ncol >> 7;
    int j     = ncol & 127;
    const float* W = (gate == 0) ? Wgh : (gate == 1) ? Wih : (gate == 2) ? Wfh : Woh;
    wh_swz[idx] = (f16)W[k * H_ + j];
  }
  if (idx < G4H_) {
    int gate = idx >> 7, j = idx & 127;
    const float* Wx = (gate == 0) ? Wgx : (gate == 1) ? Wix : (gate == 2) ? Wfx : Wox;
    const float* bb = (gate == 0) ? bg  : (gate == 1) ? bi  : (gate == 2) ? bf  : bo;
    wx_cat[idx] = Wx[j];   // I == 1, so W_*x is [1,128]
    b_cat[idx]  = bb[j];
  }
}

// ---------------------------------------------------------------------------
// Main: one WG per 16 batch rows; 8 waves; whole 512-step recurrence in-kernel.
// Wave w owns N-tiles {w, w+8, w+16, w+24} = columns [16w,16w+16) of gates
// g,i,f,o respectively -> elementwise LSTM update is wave-local.
// Wh (B fragments) is register-resident for the entire time loop.
// ---------------------------------------------------------------------------
__launch_bounds__(256, 1)
__global__ void lstm_main(const float* __restrict__ x, const f16* __restrict__ wh_swz,
                          const float* __restrict__ wx_cat, const float* __restrict__ b_cat,
                          const float* __restrict__ Wph, const float* __restrict__ bp,
                          float* __restrict__ out) {
  extern __shared__ char smem[];
  float* s_x  = (float*)(smem + L_X);    // transposed: s_x[t*16 + r]
  f16*   s_h0 = (f16*)  (smem + L_H0);
  f16*   s_h1 = (f16*)  (smem + L_H1);
  float* s_hf = (float*)(smem + L_HF);

  const int tid  = threadIdx.x;
  const int lane = tid & 31;
  const int w    = tid >> 5;          // wave 0..7
  const int b0   = blockIdx.x * 16;   // batch row base

  // ---- stage the 16x512 x tile, transposed to [t][r] ----
  for (int i = tid; i < S_ * 16; i += 256) {
    int t = i >> 4, r = i & 15;
    s_x[i] = x[(size_t)(b0 + r) * S_ + t];
  }
  // zero both h ping-pong buffers (h0 = 0)
  for (int i = tid; i < 2 * 16 * H_; i += 256) s_h0[i] = (f16)0.f;

  // ---- hoist all 16 B fragments (whole Wh) into registers, coalesced ----
  v16h bfr[16];
#pragma unroll
  for (int g = 0; g < 4; ++g) {
#pragma unroll
    for (int kt = 0; kt < 4; ++kt) {
      const f16* bb = wh_swz + (size_t)(((((w + 8 * g) * 4) + kt) * 32 + lane) << 4);
      union { v16h v; v8h h2[2]; } u;
      u.h2[0] = *(const v8h*)(bb);
      u.h2[1] = *(const v8h*)(bb + 8);
      bfr[g * 4 + kt] = u.v;
    }
  }

  // per-lane elementwise constants (8 scalars each, straight from global)
  const int jj = w * 16 + (lane & 15);     // hidden unit owned by this lane
  const int m0 = (lane >> 4) * 8;          // accumulator row base for this lane
  const float wxg = wx_cat[jj],       wxi = wx_cat[128 + jj];
  const float wxf = wx_cat[256 + jj], wxo = wx_cat[384 + jj];
  const float bg2 = b_cat[jj],        bi2 = b_cat[128 + jj];
  const float bf2 = b_cat[256 + jj],  bo2 = b_cat[384 + jj];

  float cc[8];
#pragma unroll
  for (int i = 0; i < 8; ++i) cc[i] = 0.f;

  const int arow  = lane & 15;
  const int khalf = lane >> 4;

  __syncthreads();

  for (int t = 0; t < S_; ++t) {
    const f16* s_hr = (t & 1) ? s_h1 : s_h0;   // holds h_t
    f16*       s_hw = (t & 1) ? s_h0 : s_h1;   // receives h_{t+1}

    // ---- A fragments: h tile [16x128] f16, 4 K-chunks of 32 ----
    union { v16h v; v8h h2[2]; } a[4];
#pragma unroll
    for (int kt = 0; kt < 4; ++kt) {
      const f16* ab = s_hr + arow * H_ + kt * 32 + khalf * 16;
      a[kt].h2[0] = *(const v8h*)(ab);
      a[kt].h2[1] = *(const v8h*)(ab + 8);
    }

    // ---- 4 gate tiles x 4 K-chunks of WMMA, B register-resident ----
    v8f acc[4];
#pragma unroll
    for (int g = 0; g < 4; ++g) {
      v8f z = {};
      acc[g] = z;
#pragma unroll
      for (int kt = 0; kt < 4; ++kt) {
        acc[g] = __builtin_amdgcn_wmma_f32_16x16x32_f16(
            false, a[kt].v, false, bfr[g * 4 + kt], (short)0, acc[g], false, false);
      }
    }

    // ---- per-lane x values for this step: 8 contiguous floats ----
    const float4* xp = (const float4*)(s_x + t * 16 + m0);
    const float4 xa = xp[0], xb = xp[1];
    const float xv[8] = {xa.x, xa.y, xa.z, xa.w, xb.x, xb.y, xb.z, xb.w};

    // ---- wave-local LSTM elementwise update (8 rows per lane) ----
    float hv[8];
#pragma unroll
    for (int vi = 0; vi < 8; ++vi) {
      const float gv = fast_tanh   (acc[0][vi] + xv[vi] * wxg + bg2);
      const float iv = fast_sigmoid(acc[1][vi] + xv[vi] * wxi + bi2);
      const float fv = fast_sigmoid(acc[2][vi] + xv[vi] * wxf + bf2);
      const float ov = fast_sigmoid(acc[3][vi] + xv[vi] * wxo + bo2);
      const float cn = gv * iv + cc[vi] * fv;
      cc[vi] = cn;
      hv[vi] = fast_tanh(cn) * ov;
      s_hw[(m0 + vi) * H_ + jj] = (f16)hv[vi];
    }
    if (t == S_ - 1) {
#pragma unroll
      for (int vi = 0; vi < 8; ++vi) s_hf[(m0 + vi) * H_ + jj] = hv[vi];
    }

    __syncthreads();   // h_{t+1} visible; ping-pong makes one barrier enough
  }

  // ---- tiny final projection: [16,128] @ [128,10] + b_p ----
  if (tid < 16 * C_) {
    const int r = tid / C_, c = tid % C_;
    float sum = bp[c];
    for (int k = 0; k < H_; ++k) sum += s_hf[r * H_ + k] * Wph[k * C_ + c];
    out[(size_t)(b0 + r) * C_ + c] = sum;
  }
}

// ---------------------------------------------------------------------------
extern "C" void kernel_launch(void* const* d_in, const int* in_sizes, int n_in,
                              void* d_out, int out_size, void* d_ws, size_t ws_size,
                              hipStream_t stream) {
  const float* x   = (const float*)d_in[0];
  const float* Wgx = (const float*)d_in[1];
  const float* Wgh = (const float*)d_in[2];
  const float* Wix = (const float*)d_in[3];
  const float* Wih = (const float*)d_in[4];
  const float* Wfx = (const float*)d_in[5];
  const float* Wfh = (const float*)d_in[6];
  const float* Wox = (const float*)d_in[7];
  const float* Woh = (const float*)d_in[8];
  const float* Wph = (const float*)d_in[9];
  const float* bg  = (const float*)d_in[10];
  const float* bi  = (const float*)d_in[11];
  const float* bf  = (const float*)d_in[12];
  const float* bo  = (const float*)d_in[13];
  const float* bp  = (const float*)d_in[14];

  f16*   wh_swz = (f16*)((char*)d_ws + WS_WH_OFF);
  float* wx_cat = (float*)((char*)d_ws + WS_WX_OFF);
  float* b_cat  = (float*)((char*)d_ws + WS_B_OFF);

  lstm_prep<<<256, 256, 0, stream>>>(Wgh, Wih, Wfh, Woh, Wgx, Wix, Wfx, Wox,
                                     bg, bi, bf, bo, wh_swz, wx_cat, b_cat);

  (void)hipFuncSetAttribute((const void*)lstm_main,
                            hipFuncAttributeMaxDynamicSharedMemorySize,
                            (int)SMEM_BYTES);
  lstm_main<<<B_ / 16, 256, SMEM_BYTES, stream>>>(x, wh_swz, wx_cat, b_cat,
                                                  Wph, bp, (float*)d_out);
}
